// Net_20512763805724
// MI455X (gfx1250) — matrix operand order repaired and checked by reading, not compiled, not run on hardware
//
#include <hip/hip_runtime.h>
#include <hip/hip_bf16.h>

typedef _Float16 v16h __attribute__((ext_vector_type(16)));
typedef float    v8f  __attribute__((ext_vector_type(8)));
typedef float    v2f  __attribute__((ext_vector_type(2)));
typedef int      v4i  __attribute__((ext_vector_type(4)));

#define BATCH 65536

// ---------------------------------------------------------------------------
// Prep: binarize weights to f16 (+1/-1) and swizzle into the exact per-lane
// WMMA B-operand image:  wswz[((nt*KSTEPS+ks)*32 + lane)*8 + j]  is a dword
// holding f16 pair (K=kbase+2j, K=kbase+2j+1), kbase = ks*32 + (lane>>4)*16,
// N = nt*16 + (lane&15).  Columns n >= O are zero-padded (layer 3).
// ---------------------------------------------------------------------------
__global__ void prep_weights(const float* __restrict__ w, int* __restrict__ wswz,
                             int L, int O, int KSTEPS, int total) {
  int idx = blockIdx.x * blockDim.x + threadIdx.x;
  if (idx >= total) return;
  int j    = idx & 7;
  int lane = (idx >> 3) & 31;
  int rest = idx >> 8;
  int ks   = rest % KSTEPS;
  int nt   = rest / KSTEPS;
  int n  = nt * 16 + (lane & 15);
  int kk = ks * 32 + ((lane >> 4) & 1) * 16 + 2 * j;
  union { int i; _Float16 h[2]; } u;
  u.i = 0;
  if (n < O) {
    u.h[0] = (w[(size_t)n * L + kk]     >= 0.f) ? (_Float16)1.f : (_Float16)-1.f;
    u.h[1] = (w[(size_t)n * L + kk + 1] >= 0.f) ? (_Float16)1.f : (_Float16)-1.f;
  }
  wswz[idx] = u.i;
}

__device__ __forceinline__ v16h load_b_tile(const int* sWB, int base) {
  union { v16h h; v4i i[2]; } u;
  u.i[0] = *(const v4i*)(sWB + base);
  u.i[1] = *(const v4i*)(sWB + base + 4);
  return u.h;
}

// ---------------------------------------------------------------------------
// One SLP layer:  out[b,o] = (sum(d+e^2) + s1^2 - sum(e^2)) / L^2
// s1 = e @ sign(w)^T via v_wmma_f32_16x16x32_f16.
// Block: 256 threads (8 waves). Wave: MTILES x 16 rows, NT x 16 cols.
// Weights staged in LDS once per workgroup (pre-swizzled B image).
// ---------------------------------------------------------------------------
template <int L, int NT, int OSTORE, int OSTRIDE, int MTILES>
__global__ __launch_bounds__(256)
void slp_layer_kernel(const float* __restrict__ in, const int* __restrict__ wswz,
                      float* __restrict__ out, float invL2) {
  constexpr int KSTEPS = L / 32;
  constexpr int WBDW   = NT * KSTEPS * 256;  // dwords of pre-swizzled weights

  extern __shared__ int sWB[];
  for (int i = threadIdx.x; i < WBDW; i += 256) sWB[i] = wswz[i];
  __syncthreads();

  const int lane = threadIdx.x & 31;
  const int wave = threadIdx.x >> 5;
  const int h    = lane >> 4;    // which K-half this lane covers in A
  const int mrow = lane & 15;    // A row within M-tile / C column within N-tile
  const long long rowBlock =
      (long long)blockIdx.x * (8 * MTILES * 16) + (long long)wave * (MTILES * 16);

  v8f acc[NT][MTILES];
#pragma unroll
  for (int nt = 0; nt < NT; ++nt)
#pragma unroll
    for (int mt = 0; mt < MTILES; ++mt)
#pragma unroll
      for (int k = 0; k < 8; ++k) acc[nt][mt][k] = 0.f;

  float t[MTILES], ss[MTILES];
#pragma unroll
  for (int mt = 0; mt < MTILES; ++mt) { t[mt] = 0.f; ss[mt] = 0.f; }

  for (int ks = 0; ks < KSTEPS; ++ks) {
    // ---- build A operands (contiguous K: element s holds K = 16h + s), fp32 stats on the fly
    v16h A[MTILES];
#pragma unroll
    for (int mt = 0; mt < MTILES; ++mt) {
      const float* rp =
          in + (size_t)(rowBlock + mt * 16 + mrow) * L + (size_t)ks * 32 + h * 16;
      float tAcc = 0.f, dAcc = 0.f;
#pragma unroll
      for (int j = 0; j < 8; ++j) {
        int off = 2 * j;  // K pairs {0..15} (+16h)
        v2f p = *(const v2f*)(rp + off);
        float e0 = 2.f * p[0] - 1.f;
        float e1 = 2.f * p[1] - 1.f;
        tAcc += e0 * e0 + e1 * e1;
        dAcc += 4.f * p[0] * (1.f - p[0]) + 4.f * p[1] * (1.f - p[1]);
        A[mt][2 * j]     = (_Float16)e0;
        A[mt][2 * j + 1] = (_Float16)e1;
      }
      t[mt]  += tAcc;          // s2 = sum e^2
      ss[mt] += dAcc + tAcc;   // sum(d + e^2)
    }
    // ---- B tiles from LDS, WMMA accumulate
#pragma unroll
    for (int nt = 0; nt < NT; ++nt) {
      v16h Bv = load_b_tile(sWB, ((nt * KSTEPS + ks) * 32 + lane) * 8);
#pragma unroll
      for (int mt = 0; mt < MTILES; ++mt)
        acc[nt][mt] = __builtin_amdgcn_wmma_f32_16x16x32_f16(
            false, A[mt], false, Bv, (short)0, acc[nt][mt], false, false);
    }
  }

  // combine the two K-half partials per row (wave32 xor-16)
#pragma unroll
  for (int mt = 0; mt < MTILES; ++mt) {
    t[mt]  += __shfl_xor(t[mt], 16, 32);
    ss[mt] += __shfl_xor(ss[mt], 16, 32);
  }

  // epilogue per C layout: VGPR j holds M = j + 8h, lane&15 = N
#pragma unroll
  for (int mt = 0; mt < MTILES; ++mt) {
#pragma unroll
    for (int j = 0; j < 8; ++j) {
      int rloc = j + 8 * h;
      float tr = __shfl(t[mt],  rloc, 32);
      float sr = __shfl(ss[mt], rloc, 32);
      long long grow = rowBlock + mt * 16 + rloc;
#pragma unroll
      for (int nt = 0; nt < NT; ++nt) {
        int col  = nt * 16 + mrow;
        float s1 = acc[nt][mt][j];
        float v  = (sr + s1 * s1 - tr) * invL2;
        if (col < OSTORE) out[(size_t)grow * OSTRIDE + col] = v;
      }
    }
  }
}

extern "C" void kernel_launch(void* const* d_in, const int* in_sizes, int n_in,
                              void* d_out, int out_size, void* d_ws, size_t ws_size,
                              hipStream_t stream) {
  const float* x  = (const float*)d_in[0];   // [65536, 768]
  const float* w1 = (const float*)d_in[1];   // [64, 768]
  const float* w2 = (const float*)d_in[2];   // [32, 64]
  const float* w3 = (const float*)d_in[3];   // [4, 32]
  float* out = (float*)d_out;                // [65536, 4]

  // workspace layout
  int* ws1 = (int*)d_ws;                     // 24576 dwords (96 KB)
  int* ws2 = ws1 + 24576;                    // 1024 dwords
  int* ws3 = ws2 + 1024;                     // 256 dwords
  float* h1 = (float*)(ws3 + 256);           // [65536, 64] f32 (16 MB)
  float* h2 = h1 + (size_t)BATCH * 64;       // [65536, 32] f32 (8 MB)

  // pre-swizzle binarized weights into WMMA B-operand images
  prep_weights<<<96, 256, 0, stream>>>(w1, ws1, 768, 64, 24, 24576);
  prep_weights<<<4,  256, 0, stream>>>(w2, ws2, 64, 32, 2, 1024);
  prep_weights<<<1,  256, 0, stream>>>(w3, ws3, 32, 4, 1, 256);

  // 256 rows per block -> 256 blocks per layer
  (void)hipFuncSetAttribute(
      (const void*)slp_layer_kernel<768, 4, 64, 64, 2>,
      hipFuncAttributeMaxDynamicSharedMemorySize, 24576 * 4);

  slp_layer_kernel<768, 4, 64, 64, 2>
      <<<BATCH / 256, 256, 24576 * 4, stream>>>(x, ws1, h1, 1.f / (768.f * 768.f));
  slp_layer_kernel<64, 2, 32, 32, 2>
      <<<BATCH / 256, 256, 1024 * 4, stream>>>(h1, ws2, h2, 1.f / 4096.f);
  slp_layer_kernel<32, 1, 4, 4, 2>
      <<<BATCH / 256, 256, 256 * 4, stream>>>(h2, ws3, out, 1.f / 1024.f);
}
